// Kernel_14913859789082465786_53472342835832
// MI455X (gfx1250) — matrix operand order repaired and checked by reading, not compiled, not run on hardware
//
#include <hip/hip_runtime.h>
#include <hip/hip_bf16.h>

typedef __bf16 v16bf __attribute__((ext_vector_type(16)));
typedef __bf16 v8bf  __attribute__((ext_vector_type(8)));
typedef float  v8f   __attribute__((ext_vector_type(8)));

#define C_DIM 256
#define S_DIM 4096   // H*W = 64*64
#define N_DIM 16

// ---------------------------------------------------------------------------
// WMMA fragment helpers (CDNA5 wave32 layouts, 16-bit operands, 16x16x32)
//   A (16x32): lane l -> M = l%16 ; half = l/16
//              e=0..7  -> K = 8*half + e        (contiguous 8)
//              e=8..15 -> K = 16 + 8*half + e-8 (contiguous 8)
//   B (32x16): lane l -> N = l%16 ; K same as A (8*half + e, 16 + 8*half + e-8)
//   C/D f32 (16x16): vgpr r -> M = r + 8*half ; N = l%16
// ---------------------------------------------------------------------------

__device__ __forceinline__ v16bf frag_zero() {
  v16bf r;
#pragma unroll
  for (int e = 0; e < 16; ++e) r[e] = (__bf16)0.0f;
  return r;
}

// rowAbs: absolute M row (already includes lane%16)
__device__ __forceinline__ v16bf load_frag_a(const __bf16* __restrict__ base,
                                             int ld, int rowAbs, int kb, int half) {
  const __bf16* p = base + (size_t)rowAbs * ld + kb + 8 * half;
  v8bf lo = *(const v8bf*)p;
  v8bf hi = *(const v8bf*)(p + 16);
  v16bf r;
#pragma unroll
  for (int e = 0; e < 8; ++e) { r[e] = lo[e]; r[e + 8] = hi[e]; }
  return r;
}

// rowAbs: absolute N row (already includes lane%16)
__device__ __forceinline__ v16bf load_frag_b(const __bf16* __restrict__ base,
                                             int ld, int rowAbs, int kb, int half) {
  const __bf16* p = base + (size_t)rowAbs * ld + kb + 8 * half;
  v8bf lo = *(const v8bf*)p;
  v8bf hi = *(const v8bf*)(p + 16);
  v16bf r;
#pragma unroll
  for (int e = 0; e < 8; ++e) { r[e] = lo[e]; r[e + 8] = hi[e]; }
  return r;
}

// ---------------------------------------------------------------------------
// Kernel 1: pack conv weights [C,C,1,5] f32 -> Wpack[dw][o][i] bf16 (K-fast)
// ---------------------------------------------------------------------------
__global__ void __launch_bounds__(256)
pack_w_kernel(const float* __restrict__ p3w, __bf16* __restrict__ wpack) {
  int idx = blockIdx.x * 256 + threadIdx.x;          // < 5*256*256
  if (idx >= 5 * C_DIM * C_DIM) return;
  int dw  = idx >> 16;                               // /65536
  int rem = idx & 0xFFFF;                            // o*256+i
  wpack[idx] = (__bf16)p3w[(size_t)rem * 5 + dw];
}

// ---------------------------------------------------------------------------
// Kernel 2: t6[n][c][s] = sum_k p4[c,k]*p6[k]*x[n,c,h,w+k-3]  (1x7 depthwise)
// ---------------------------------------------------------------------------
__global__ void __launch_bounds__(256)
prep_t6_kernel(const float* __restrict__ x, const float* __restrict__ p4,
               const float* __restrict__ p6, __bf16* __restrict__ t6) {
  size_t idx = (size_t)blockIdx.x * 256 + threadIdx.x;   // < 16*256*4096
  int w = (int)(idx & 63);
  int c = (int)((idx >> 12) & 255);
  const float* row = x + (idx & ~(size_t)63);
  float acc = 0.0f;
#pragma unroll
  for (int k = 0; k < 7; ++k) {
    int ww = w + k - 3;
    if (ww >= 0 && ww < 64) acc += p4[c * 7 + k] * p6[k] * row[ww];
  }
  t6[idx] = (__bf16)acc;
}

// ---------------------------------------------------------------------------
// Kernel 3: LDS-tiled transpose producing
//   t2T[n][s][c] = bf16(p2w[c][h] * x[n][c][s])
//   t5T[n][s][c] = bf16(relu(x[n][c][s]))
// grid: (N, C/32, S/32), block 256
// ---------------------------------------------------------------------------
__global__ void __launch_bounds__(256)
transpose_prep_kernel(const float* __restrict__ x, const float* __restrict__ p2w,
                      __bf16* __restrict__ t2T, __bf16* __restrict__ t5T) {
  __shared__ float tile[32][33];
  const int n  = blockIdx.x;
  const int ct = blockIdx.y * 32;
  const int st = blockIdx.z * 32;
  const int j  = threadIdx.x & 31;
  const int i0 = threadIdx.x >> 5;       // 0..7
  const float* xb = x + (size_t)n * C_DIM * S_DIM;
#pragma unroll
  for (int r = 0; r < 4; ++r) {
    int i = i0 + 8 * r;                  // local c
    tile[i][j] = xb[(size_t)(ct + i) * S_DIM + st + j];
  }
  __syncthreads();
#pragma unroll
  for (int r = 0; r < 4; ++r) {
    int i = i0 + 8 * r;                  // local s
    int s = st + i;
    int c = ct + j;
    float v  = tile[j][i];
    int h    = s >> 6;
    float sc = p2w[c * 64 + h];
    size_t o = ((size_t)n * S_DIM + s) * C_DIM + c;
    t2T[o] = (__bf16)(sc * v);
    t5T[o] = (__bf16)(v > 0.0f ? v : 0.0f);
  }
}

// ---------------------------------------------------------------------------
// Kernel 4: conv as 5 shifted NT-GEMM passes
//   t3[n][o][s] = sum_dw sum_i Wpack[dw][o][i] * t2T[n][s+dw-2][i]
// grid (N, 2, 32), block 128 (4 waves, 2x2 of 64x64 tiles)
// ---------------------------------------------------------------------------
__global__ void __launch_bounds__(128)
gemm_conv_kernel(const __bf16* __restrict__ wpack, const __bf16* __restrict__ t2T,
                 __bf16* __restrict__ t3) {
  const int n    = blockIdx.x;
  const int lane = threadIdx.x & 31;
  const int wid  = threadIdx.x >> 5;
  const int l16  = lane & 15;
  const int half = (lane >> 4) & 1;
  const int mwave = blockIdx.y * 128 + (wid >> 1) * 64;   // o
  const int nwave = blockIdx.z * 128 + (wid & 1) * 64;    // s
  const __bf16* Bt = t2T + (size_t)n * S_DIM * C_DIM;

  v8f acc[4][4];
#pragma unroll
  for (int i = 0; i < 4; ++i)
#pragma unroll
    for (int j = 0; j < 4; ++j)
#pragma unroll
      for (int e = 0; e < 8; ++e) acc[i][j][e] = 0.0f;

  for (int dw = 0; dw < 5; ++dw) {
    const __bf16* A = wpack + (size_t)dw * C_DIM * C_DIM;
    int  srow[4];
    bool valid[4];
#pragma unroll
    for (int j = 0; j < 4; ++j) {
      int s  = nwave + 16 * j + l16;
      int ws = (s & 63) + dw - 2;
      valid[j] = (ws >= 0) && (ws < 64);
      srow[j]  = s + dw - 2;
    }
    for (int kb = 0; kb < C_DIM; kb += 32) {
      v16bf fa[4], fb[4];
#pragma unroll
      for (int i = 0; i < 4; ++i)
        fa[i] = load_frag_a(A, C_DIM, mwave + 16 * i + l16, kb, half);
#pragma unroll
      for (int j = 0; j < 4; ++j) {
        if (valid[j]) fb[j] = load_frag_b(Bt, C_DIM, srow[j], kb, half);
        else          fb[j] = frag_zero();
      }
      // prefetch next K-tile (global_prefetch_b8; no counter traffic)
      if (kb + 64 < C_DIM) {
#pragma unroll
        for (int i = 0; i < 4; ++i)
          __builtin_prefetch(A + (size_t)(mwave + 16 * i + l16) * C_DIM + kb + 64, 0, 3);
#pragma unroll
        for (int j = 0; j < 4; ++j)
          if (valid[j])
            __builtin_prefetch(Bt + (size_t)srow[j] * C_DIM + kb + 64, 0, 3);
      }
#pragma unroll
      for (int i = 0; i < 4; ++i)
#pragma unroll
        for (int j = 0; j < 4; ++j)
          acc[i][j] = __builtin_amdgcn_wmma_f32_16x16x32_bf16(
              false, fa[i], false, fb[j], (short)0, acc[i][j], false, false);
    }
  }

  __bf16* D = t3 + (size_t)n * C_DIM * S_DIM;   // [o][s]
#pragma unroll
  for (int i = 0; i < 4; ++i)
#pragma unroll
    for (int j = 0; j < 4; ++j) {
      int s = nwave + 16 * j + l16;
#pragma unroll
      for (int r = 0; r < 8; ++r) {
        int o = mwave + 16 * i + r + 8 * half;
        D[(size_t)o * S_DIM + s] = (__bf16)acc[i][j][r];
      }
    }
}

// ---------------------------------------------------------------------------
// Kernel 5: t7T[n][d][c] = (1/64) * sum_s t3[n][c][s] * t6[n][d][s]   (NT)
// grid (N, 2, 2), block 128
// ---------------------------------------------------------------------------
__global__ void __launch_bounds__(128)
gemm_t7_kernel(const __bf16* __restrict__ t3, const __bf16* __restrict__ t6,
               __bf16* __restrict__ t7T) {
  const int n    = blockIdx.x;
  const int lane = threadIdx.x & 31;
  const int wid  = threadIdx.x >> 5;
  const int l16  = lane & 15;
  const int half = (lane >> 4) & 1;
  const int mwave = blockIdx.y * 128 + (wid >> 1) * 64;   // c
  const int nwave = blockIdx.z * 128 + (wid & 1) * 64;    // d
  const __bf16* A = t3 + (size_t)n * C_DIM * S_DIM;
  const __bf16* B = t6 + (size_t)n * C_DIM * S_DIM;

  v8f acc[4][4];
#pragma unroll
  for (int i = 0; i < 4; ++i)
#pragma unroll
    for (int j = 0; j < 4; ++j)
#pragma unroll
      for (int e = 0; e < 8; ++e) acc[i][j][e] = 0.0f;

  for (int kb = 0; kb < S_DIM; kb += 32) {
    v16bf fa[4], fb[4];
#pragma unroll
    for (int i = 0; i < 4; ++i)
      fa[i] = load_frag_a(A, S_DIM, mwave + 16 * i + l16, kb, half);
#pragma unroll
    for (int j = 0; j < 4; ++j)
      fb[j] = load_frag_b(B, S_DIM, nwave + 16 * j + l16, kb, half);
    if (kb + 64 < S_DIM) {
#pragma unroll
      for (int i = 0; i < 4; ++i)
        __builtin_prefetch(A + (size_t)(mwave + 16 * i + l16) * S_DIM + kb + 64, 0, 3);
#pragma unroll
      for (int j = 0; j < 4; ++j)
        __builtin_prefetch(B + (size_t)(nwave + 16 * j + l16) * S_DIM + kb + 64, 0, 3);
    }
#pragma unroll
    for (int i = 0; i < 4; ++i)
#pragma unroll
      for (int j = 0; j < 4; ++j)
        acc[i][j] = __builtin_amdgcn_wmma_f32_16x16x32_bf16(
            false, fa[i], false, fb[j], (short)0, acc[i][j], false, false);
  }

  __bf16* D = t7T + (size_t)n * C_DIM * C_DIM;   // [d][c]
#pragma unroll
  for (int i = 0; i < 4; ++i)
#pragma unroll
    for (int j = 0; j < 4; ++j) {
      int d = nwave + 16 * j + l16;
#pragma unroll
      for (int r = 0; r < 8; ++r) {
        int c = mwave + 16 * i + r + 8 * half;
        D[(size_t)d * C_DIM + c] = (__bf16)(acc[i][j][r] * 0.015625f);  // 1/sqrt(4096)
      }
    }
}

// ---------------------------------------------------------------------------
// Kernel 6: out[n][d][s] = (1/16) * sum_c t7T[n][d][c] * t5T[n][s][c]  (NT)
// grid (N, 2, 32), block 128. Output fp32.
// ---------------------------------------------------------------------------
__global__ void __launch_bounds__(128)
gemm_out_kernel(const __bf16* __restrict__ t7T, const __bf16* __restrict__ t5T,
                float* __restrict__ out) {
  const int n    = blockIdx.x;
  const int lane = threadIdx.x & 31;
  const int wid  = threadIdx.x >> 5;
  const int l16  = lane & 15;
  const int half = (lane >> 4) & 1;
  const int mwave = blockIdx.y * 128 + (wid >> 1) * 64;   // d
  const int nwave = blockIdx.z * 128 + (wid & 1) * 64;    // s
  const __bf16* A = t7T + (size_t)n * C_DIM * C_DIM;
  const __bf16* B = t5T + (size_t)n * S_DIM * C_DIM;

  v8f acc[4][4];
#pragma unroll
  for (int i = 0; i < 4; ++i)
#pragma unroll
    for (int j = 0; j < 4; ++j)
#pragma unroll
      for (int e = 0; e < 8; ++e) acc[i][j][e] = 0.0f;

  for (int kb = 0; kb < C_DIM; kb += 32) {
    v16bf fa[4], fb[4];
#pragma unroll
    for (int i = 0; i < 4; ++i)
      fa[i] = load_frag_a(A, C_DIM, mwave + 16 * i + l16, kb, half);
#pragma unroll
    for (int j = 0; j < 4; ++j)
      fb[j] = load_frag_b(B, C_DIM, nwave + 16 * j + l16, kb, half);
    if (kb + 64 < C_DIM) {
#pragma unroll
      for (int i = 0; i < 4; ++i)
        __builtin_prefetch(A + (size_t)(mwave + 16 * i + l16) * C_DIM + kb + 64, 0, 3);
#pragma unroll
      for (int j = 0; j < 4; ++j)
        __builtin_prefetch(B + (size_t)(nwave + 16 * j + l16) * C_DIM + kb + 64, 0, 3);
    }
#pragma unroll
    for (int i = 0; i < 4; ++i)
#pragma unroll
      for (int j = 0; j < 4; ++j)
        acc[i][j] = __builtin_amdgcn_wmma_f32_16x16x32_bf16(
            false, fa[i], false, fb[j], (short)0, acc[i][j], false, false);
  }

  float* On = out + (size_t)n * C_DIM * S_DIM;   // [d][s]
#pragma unroll
  for (int i = 0; i < 4; ++i)
#pragma unroll
    for (int j = 0; j < 4; ++j) {
      int s = nwave + 16 * j + l16;
#pragma unroll
      for (int r = 0; r < 8; ++r) {
        int d = mwave + 16 * i + r + 8 * half;
        On[(size_t)d * S_DIM + s] = acc[i][j][r] * 0.0625f;   // 1/sqrt(256)
      }
    }
}

// ---------------------------------------------------------------------------
extern "C" void kernel_launch(void* const* d_in, const int* in_sizes, int n_in,
                              void* d_out, int out_size, void* d_ws, size_t ws_size,
                              hipStream_t stream) {
  const float* x   = (const float*)d_in[0];   // [16,256,64,64]
  const float* p2w = (const float*)d_in[1];   // [1,256,64,1]
  const float* p3w = (const float*)d_in[2];   // [256,256,1,5]
  const float* p4w = (const float*)d_in[3];   // [1,256,7,1,1]
  const float* p6w = (const float*)d_in[4];   // [7,1]
  float* out = (float*)d_out;

  const size_t big = (size_t)N_DIM * C_DIM * S_DIM;   // 16.78M elems
  char* ws = (char*)d_ws;
  __bf16* t2T   = (__bf16*)(ws);
  __bf16* t5T   = (__bf16*)(ws + 2 * big);
  __bf16* t6    = (__bf16*)(ws + 4 * big);
  __bf16* t3    = (__bf16*)(ws + 6 * big);
  __bf16* t7T   = (__bf16*)(ws + 8 * big);
  __bf16* wpack = (__bf16*)(ws + 8 * big + 2 * (size_t)N_DIM * C_DIM * C_DIM);

  pack_w_kernel<<<(5 * C_DIM * C_DIM + 255) / 256, 256, 0, stream>>>(p3w, wpack);
  prep_t6_kernel<<<(unsigned)(big / 256), 256, 0, stream>>>(x, p4w, p6w, t6);
  dim3 gT(N_DIM, C_DIM / 32, S_DIM / 32);
  transpose_prep_kernel<<<gT, 256, 0, stream>>>(x, p2w, t2T, t5T);
  dim3 g2(N_DIM, C_DIM / 128, S_DIM / 128);
  gemm_conv_kernel<<<g2, 128, 0, stream>>>(wpack, t2T, t3);
  dim3 g3(N_DIM, C_DIM / 128, C_DIM / 128);
  gemm_t7_kernel<<<g3, 128, 0, stream>>>(t3, t6, t7T);
  dim3 g4(N_DIM, C_DIM / 128, S_DIM / 128);
  gemm_out_kernel<<<g4, 128, 0, stream>>>(t7T, t5T, out);
}
